// GraphConvBlockI_9740985827428
// MI455X (gfx1250) — matrix operand order repaired and checked by reading, not compiled, not run on hardware
//
#include <hip/hip_runtime.h>

typedef __attribute__((ext_vector_type(16))) __bf16 v16bf;
typedef __attribute__((ext_vector_type(8)))  __bf16 v8bf;
typedef __attribute__((ext_vector_type(8)))  float  v8f;
typedef __attribute__((ext_vector_type(4)))  float  f32x4;
typedef __attribute__((ext_vector_type(2)))  unsigned int u32x2;

#define BETA   0.2f
#define JNT    17
#define CH     128
#define SAMP   16                 // batch samples per workgroup
#define ROWS   (SAMP * JNT)       // 272 rows = 17 m-tiles of 16
#define MT     (ROWS / 16)        // 17
#define XSTR   132                // padded f32 row stride (bank spread, 16B aligned)
#define ASTR   136                // padded bf16 row stride (16B aligned)
#define NLAYER 5

union FragBF { v16bf v; v8bf h[2]; };
union Pack4  { __bf16 b[4]; u32x2 u; };

__global__ __launch_bounds__(256) void gconv5_ln_kernel(
    const float* __restrict__ xg,  const float* __restrict__ x0g,
    const float* __restrict__ adj, const float* __restrict__ W0,
    const float* __restrict__ W1,  const float* __restrict__ bias,
    const float* __restrict__ gam, const float* __restrict__ bet,
    float* __restrict__ outg)
{
    __shared__ float  sX[ROWS * XSTR];        // f32 activations (residual path)
    __shared__ __bf16 sXa[ROWS * ASTR];       // bf16 neighbor-aggregated activations
    __shared__ __bf16 sW0t[CH * ASTR];        // W0^T bf16: [n][k]
    __shared__ __bf16 sW1t[CH * ASTR];        // W1^T bf16
    __shared__ float  sAdjT[JNT * 20];        // sAdjT[k*20+j] = adj_off[j][k], padded
    __shared__ float  sDiagRow[ROWS];         // adj diag per row
    __shared__ float  sBias[CH];

    const int tid  = threadIdx.x;
    const int wv   = tid >> 5;                // wave id == N-tile id (8 N-tiles)
    const int lane = tid & 31;
    const int ln16 = lane & 15;
    const int hi   = lane >> 4;               // 0: lanes 0-15, 1: lanes 16-31
    const long Rbase = (long)blockIdx.x * ROWS;

    // ---- async DMA: x tile -> LDS (HBM -> LDS direct, ASYNCcnt-tracked) ----
    {
        const float* xt = xg + Rbase * CH;
        for (int idx = tid; idx < ROWS * (CH / 4); idx += 256) {
            int r = idx >> 5, c4 = idx & 31;
            unsigned lds = (unsigned)(size_t)&sX[r * XSTR + c4 * 4];
            const float* ga = xt + idx * 4;
            asm volatile("global_load_async_to_lds_b128 %0, %1, off"
                         :: "v"(lds), "v"(ga) : "memory");
        }
    }
    // ---- meanwhile: weights -> bf16 transposed; adjacency; x0 prefetch ----
    for (int e = tid; e < CH * CH; e += 256) {
        int k = e >> 7, n = e & 127;          // W[k][n] row-major, coalesced read
        sW0t[n * ASTR + k] = (__bf16)W0[e];
        sW1t[n * ASTR + k] = (__bf16)W1[e];
    }
    for (int e = tid; e < JNT * JNT; e += 256) {
        int j = e / JNT, k = e - j * JNT;
        sAdjT[k * 20 + j] = (j == k) ? 0.0f : adj[e];   // transposed off-diag
    }
    for (int r = tid; r < ROWS; r += 256) {
        int j = r % JNT;
        sDiagRow[r] = adj[j * JNT + j];
    }
    if (tid < CH) sBias[tid] = bias[tid];
    {   // warm L2/L0 for the layer-0 epilogue's strided x0 reads
        const float* x0t = x0g + Rbase * CH;
        for (int cl = tid; cl < (ROWS * CH) / 64; cl += 256)
            __builtin_prefetch(x0t + cl * 64, 0, 1);
    }
    asm volatile("s_wait_asynccnt 0" ::: "memory");
    __syncthreads();

    const float b_lane = sBias[wv * 16 + ln16];
    v8f acc[MT];

    for (int L = 0; L < NLAYER; ++L) {
        // ---- Phase A: Xa = A_off @ X. One thread owns (sample, c4) slab and
        //      produces all 17 joint outputs: sources loaded once, k-outer. ----
        {
            const int s0 = tid >> 5, c4 = tid & 31;   // tasks: (s0,c4) and (s0+8,c4)
            const int s1 = s0 + 8;
            f32x4 a0[JNT], a1[JNT];
            #pragma unroll
            for (int j = 0; j < JNT; ++j) {
                f32x4 z = {0.f, 0.f, 0.f, 0.f};
                a0[j] = z; a1[j] = z;
            }
            #pragma unroll 1
            for (int k = 0; k < JNT; ++k) {
                f32x4 xv0 = *(const f32x4*)&sX[(s0 * JNT + k) * XSTR + c4 * 4];
                f32x4 xv1 = *(const f32x4*)&sX[(s1 * JNT + k) * XSTR + c4 * 4];
                const f32x4* arow = (const f32x4*)&sAdjT[k * 20];  // broadcast addr
                f32x4 w0 = arow[0], w1 = arow[1], w2 = arow[2], w3 = arow[3], w4 = arow[4];
                #pragma unroll
                for (int j = 0; j < JNT; ++j) {
                    float w = (j < 4)  ? w0[j]      : (j < 8)  ? w1[j - 4]  :
                              (j < 12) ? w2[j - 8]  : (j < 16) ? w3[j - 12] : w4[0];
                    a0[j] += w * xv0;
                    a1[j] += w * xv1;
                }
            }
            #pragma unroll
            for (int j = 0; j < JNT; ++j) {
                Pack4 p0, p1;
                p0.b[0] = (__bf16)a0[j][0]; p0.b[1] = (__bf16)a0[j][1];
                p0.b[2] = (__bf16)a0[j][2]; p0.b[3] = (__bf16)a0[j][3];
                p1.b[0] = (__bf16)a1[j][0]; p1.b[1] = (__bf16)a1[j][1];
                p1.b[2] = (__bf16)a1[j][2]; p1.b[3] = (__bf16)a1[j][3];
                *(u32x2*)&sXa[(s0 * JNT + j) * ASTR + c4 * 4] = p0.u;
                *(u32x2*)&sXa[(s1 * JNT + j) * ASTR + c4 * 4] = p1.u;
            }
        }
        __syncthreads();

        // ---- Phase B: acc = (diag*X)@W0 + Xa@W1, bf16 WMMA, f32 accumulate ----
        #pragma unroll
        for (int m = 0; m < MT; ++m) {
            v8f z = {0.f, 0.f, 0.f, 0.f, 0.f, 0.f, 0.f, 0.f};
            acc[m] = z;
        }
        const int nb = wv * 16 + ln16;        // B-matrix column owned by this lane
        #pragma unroll
        for (int kb = 0; kb < 4; ++kb) {
            // B fragment: lane<16 holds K=kb*32+{0..7,16..23}, lane>=16 holds +8
            const int koB = kb * 32 + hi * 8;
            FragBF B0f, B1f;
            B0f.h[0] = *(const v8bf*)&sW0t[nb * ASTR + koB];
            B0f.h[1] = *(const v8bf*)&sW0t[nb * ASTR + koB + 16];
            B1f.h[0] = *(const v8bf*)&sW1t[nb * ASTR + koB];
            B1f.h[1] = *(const v8bf*)&sW1t[nb * ASTR + koB + 16];
            // A layout: lane<16 -> K {klo..klo+7, klo+16..+23}, klo=kb*32 (+8 hi half)
            const int klo = kb * 32 + hi * 8;
            const int khi = klo + 16;
            #pragma unroll
            for (int m = 0; m < MT; ++m) {
                const int r = m * 16 + ln16;
                const float d = sDiagRow[r];
                const float* xr = &sX[r * XSTR];
                f32x4 p0 = *(const f32x4*)&xr[klo];
                f32x4 p1 = *(const f32x4*)&xr[klo + 4];
                f32x4 q0 = *(const f32x4*)&xr[khi];
                f32x4 q1 = *(const f32x4*)&xr[khi + 4];
                FragBF A0;
                A0.v[0]  = (__bf16)(d * p0[0]); A0.v[1]  = (__bf16)(d * p0[1]);
                A0.v[2]  = (__bf16)(d * p0[2]); A0.v[3]  = (__bf16)(d * p0[3]);
                A0.v[4]  = (__bf16)(d * p1[0]); A0.v[5]  = (__bf16)(d * p1[1]);
                A0.v[6]  = (__bf16)(d * p1[2]); A0.v[7]  = (__bf16)(d * p1[3]);
                A0.v[8]  = (__bf16)(d * q0[0]); A0.v[9]  = (__bf16)(d * q0[1]);
                A0.v[10] = (__bf16)(d * q0[2]); A0.v[11] = (__bf16)(d * q0[3]);
                A0.v[12] = (__bf16)(d * q1[0]); A0.v[13] = (__bf16)(d * q1[1]);
                A0.v[14] = (__bf16)(d * q1[2]); A0.v[15] = (__bf16)(d * q1[3]);
                FragBF A1;
                A1.h[0] = *(const v8bf*)&sXa[r * ASTR + klo];
                A1.h[1] = *(const v8bf*)&sXa[r * ASTR + khi];
                acc[m] = __builtin_amdgcn_wmma_f32_16x16x32_bf16(
                    false, A0.v, false, B0f.v, (short)0, acc[m], false, false);
                acc[m] = __builtin_amdgcn_wmma_f32_16x16x32_bf16(
                    false, A1.v, false, B1f.v, (short)0, acc[m], false, false);
            }
        }
        __syncthreads();  // all waves done reading sX/sXa for this layer

        // ---- Phase C: residual epilogue, f32, in-place into sX ----
        {
            const int col = wv * 16 + ln16;
            #pragma unroll
            for (int m = 0; m < MT; ++m) {
                const int rb = m * 16 + hi * 8;   // D layout: VGPR v -> row v + 8*hi
                #pragma unroll
                for (int v = 0; v < 8; ++v) {
                    const int r = rb + v;
                    float xo = x0g[(Rbase + r) * CH + col];
                    sX[r * XSTR + col] = BETA * xo + (1.0f - BETA) * acc[m][v] + b_lane;
                }
            }
        }
        __syncthreads();
    }

    // ---- LayerNorm: one wave per row, wave32 shuffle reductions, coalesced store ----
    {
        const f32x4 g4 = ((const f32x4*)gam)[lane];
        const f32x4 e4 = ((const f32x4*)bet)[lane];
        for (int r = wv; r < ROWS; r += 8) {
            f32x4 xv = *(const f32x4*)&sX[r * XSTR + lane * 4];
            float s = xv[0] + xv[1] + xv[2] + xv[3];
            #pragma unroll
            for (int o = 16; o > 0; o >>= 1) s += __shfl_xor(s, o, 32);
            const float mu = s * (1.0f / 128.0f);
            f32x4 dv = xv - mu;
            float ss = dv[0]*dv[0] + dv[1]*dv[1] + dv[2]*dv[2] + dv[3]*dv[3];
            #pragma unroll
            for (int o = 16; o > 0; o >>= 1) ss += __shfl_xor(ss, o, 32);
            const float rs = rsqrtf(ss * (1.0f / 128.0f) + 1e-10f);
            f32x4 ov = dv * rs * g4 + e4;
            *(f32x4*)&outg[(Rbase + r) * CH + lane * 4] = ov;
        }
    }
}

extern "C" void kernel_launch(void* const* d_in, const int* in_sizes, int n_in,
                              void* d_out, int out_size, void* d_ws, size_t ws_size,
                              hipStream_t stream) {
    const float* x    = (const float*)d_in[0];
    const float* x0   = (const float*)d_in[1];
    const float* adj  = (const float*)d_in[2];
    const float* W0   = (const float*)d_in[3];
    const float* W1   = (const float*)d_in[4];
    const float* b    = (const float*)d_in[5];
    const float* gam  = (const float*)d_in[6];
    const float* bet  = (const float*)d_in[7];
    const int B = in_sizes[0] / (JNT * CH);       // 16384
    const int nblocks = B / SAMP;                 // 1024
    hipLaunchKernelGGL(gconv5_ln_kernel, dim3(nblocks), dim3(256), 0, stream,
                       x, x0, adj, W0, W1, b, gam, bet, (float*)d_out);
}
